// CrossAttentionBlock_2035814498318
// MI455X (gfx1250) — matrix operand order repaired and checked by reading, not compiled, not run on hardware
//
#include <hip/hip_runtime.h>
#include <hip/hip_bf16.h>

// ---------------------------------------------------------------------------
// CrossAttentionBlock on MI455X (gfx1250, wave32, WMMA bf16 w/ f32 accumulate)
// B=16384, C=1024, D=128
// ---------------------------------------------------------------------------

#define B_ROWS 16384
#define C_DIM  1024
#define D_DIM  128
#define LDK    40     // padded LDS stride (bf16 elems) for 32-wide K tiles
#define LDY    136    // padded LDS stride (bf16 elems) for 128-wide y tile

typedef __attribute__((ext_vector_type(16))) __bf16 v16bf;
typedef __attribute__((ext_vector_type(8)))  float  v8f;

// fp32 -> bf16 round-to-nearest-even (bit-exact, no libcalls)
__device__ __forceinline__ __bf16 f2bf(float f) {
  unsigned u = __builtin_bit_cast(unsigned, f);
  u += 0x7FFFu + ((u >> 16) & 1u);
  unsigned short h = (unsigned short)(u >> 16);
  return __builtin_bit_cast(__bf16, h);
}

__device__ __forceinline__ v8f wmma_bf16(v16bf a, v16bf b, v8f c) {
  // (neg_a, A, neg_b, B, c_mod, C, reuse_a, reuse_b)
  return __builtin_amdgcn_wmma_f32_16x16x32_bf16(false, a, false, b,
                                                 (short)0, c, false, false);
}

// A-matrix 16x32 bf16 fragment from row-major LDS tile (ISA 7.12.2):
//   lanes 0-15 : row M=lane,   elems 0-7 -> K 0..7,  elems 8-15 -> K 16..23
//   lanes 16-31: row M=lane-16,elems 0-7 -> K 8..15, elems 8-15 -> K 24..31
__device__ __forceinline__ v16bf load_a_frag(const __bf16* tile, int ldk) {
  const int l  = threadIdx.x & 31;
  const int m  = l & 15;
  const int kh = (l >> 4) * 8;
  const __bf16* row = tile + m * ldk + kh;
  v16bf a;
#pragma unroll
  for (int i = 0; i < 8; ++i) a[i] = row[i];
#pragma unroll
  for (int i = 0; i < 8; ++i) a[8 + i] = row[16 + i];
  return a;
}

// B-matrix 32x16 bf16 fragment; LDS tile stored [n][k] row-major:
//   lanes 0-15 : col N=lane,    elems 0-7 -> K 0..7,  elems 8-15 -> K 16..23
//   lanes 16-31: col N=lane-16, elems 0-7 -> K 8..15, elems 8-15 -> K 24..31
__device__ __forceinline__ v16bf load_b_frag(const __bf16* tile, int ldk) {
  const int l  = threadIdx.x & 31;
  const int n  = l & 15;
  const int k0 = (l >> 4) * 8;
  const __bf16* p = tile + n * ldk + k0;
  v16bf b;
#pragma unroll
  for (int i = 0; i < 8; ++i) b[i] = p[i];
#pragma unroll
  for (int i = 0; i < 8; ++i) b[8 + i] = p[16 + i];
  return b;
}

__device__ __forceinline__ v8f v8f_zero() {
  v8f z;
#pragma unroll
  for (int i = 0; i < 8; ++i) z[i] = 0.0f;
  return z;
}

// ---------------------------------------------------------------------------
// Kernel 1: three fused projections.  gx = x0@g_w^T+g_b ; thx/phx from x1.
// Block = 256 thr (8 waves), tile = 64 rows x 128 cols, grid = B/64 = 256.
// wave w: row-tile rt=w&3 (16 rows), col-half nh=w>>2 (64 cols) -> 12 WMMA/K.
// ---------------------------------------------------------------------------
__global__ __launch_bounds__(256) void proj_kernel(
    const float* __restrict__ x0, const float* __restrict__ x1,
    const float* __restrict__ g_w,  const float* __restrict__ g_b,
    const float* __restrict__ th_w, const float* __restrict__ th_b,
    const float* __restrict__ ph_w, const float* __restrict__ ph_b,
    float* __restrict__ gx, float* __restrict__ thx, float* __restrict__ phx) {
  __shared__ __bf16 sx0[64 * LDK];
  __shared__ __bf16 sx1[64 * LDK];
  __shared__ __bf16 swg[128 * LDK];
  __shared__ __bf16 swt[128 * LDK];
  __shared__ __bf16 swp[128 * LDK];

  const int t  = threadIdx.x;
  const int l  = t & 31;
  const int w  = t >> 5;
  const int rt = w & 3;   // 16-row tile within block
  const int nh = w >> 2;  // 0/1 -> output cols [0,64) / [64,128)
  const int m0 = blockIdx.x * 64;

  v8f accg[4], acct[4], accp[4];
#pragma unroll
  for (int nt = 0; nt < 4; ++nt) {
    accg[nt] = v8f_zero(); acct[nt] = v8f_zero(); accp[nt] = v8f_zero();
  }

  // cooperative-load coordinates
  const int xr = t >> 2;          // 0..63  (x-tile row)
  const int xk = (t & 3) * 8;     // 8 f32 per thread per x tile
  const int wr = t >> 1;          // 0..127 (weight-tile row = out col n)
  const int wk = (t & 1) * 16;    // 16 f32 per thread per weight tile

  for (int kc = 0; kc < C_DIM; kc += 32) {
    __syncthreads();
    {
      const float* p0 = x0 + (size_t)(m0 + xr) * C_DIM + kc + xk;
      const float* p1 = x1 + (size_t)(m0 + xr) * C_DIM + kc + xk;
#pragma unroll
      for (int i = 0; i < 8; ++i) {
        sx0[xr * LDK + xk + i] = f2bf(p0[i]);
        sx1[xr * LDK + xk + i] = f2bf(p1[i]);
      }
      const float* pg = g_w  + (size_t)wr * C_DIM + kc + wk;
      const float* pt = th_w + (size_t)wr * C_DIM + kc + wk;
      const float* pp = ph_w + (size_t)wr * C_DIM + kc + wk;
#pragma unroll
      for (int i = 0; i < 16; ++i) {
        swg[wr * LDK + wk + i] = f2bf(pg[i]);
        swt[wr * LDK + wk + i] = f2bf(pt[i]);
        swp[wr * LDK + wk + i] = f2bf(pp[i]);
      }
    }
    __syncthreads();

    v16bf a0 = load_a_frag(sx0 + rt * 16 * LDK, LDK);
    v16bf a1 = load_a_frag(sx1 + rt * 16 * LDK, LDK);
#pragma unroll
    for (int nt = 0; nt < 4; ++nt) {
      const int nb = (nh * 64 + nt * 16) * LDK;
      accg[nt] = wmma_bf16(a0, load_b_frag(swg + nb, LDK), accg[nt]);
      acct[nt] = wmma_bf16(a1, load_b_frag(swt + nb, LDK), acct[nt]);
      accp[nt] = wmma_bf16(a1, load_b_frag(swp + nb, LDK), accp[nt]);
    }
  }

  // C/D layout: VGPR r -> row r + (lane>=16 ? 8 : 0), col = lane&15
  const int colb = nh * 64 + (l & 15);
  const int rowb = m0 + rt * 16 + ((l >> 4) * 8);
#pragma unroll
  for (int nt = 0; nt < 4; ++nt) {
    const int col = colb + nt * 16;
    const float bg = g_b[col], bt = th_b[col], bp = ph_b[col];
#pragma unroll
    for (int r = 0; r < 8; ++r) {
      const size_t o = (size_t)(rowb + r) * D_DIM + col;
      gx[o]  = accg[nt][r] + bg;
      thx[o] = acct[nt][r] + bt;
      phx[o] = accp[nt][r] + bp;
    }
  }
}

// ---------------------------------------------------------------------------
// Kernel 2: rank-1 softmax attention, never materializing [B,128,128].
// One wave per row; lane handles i in {l, l+32, l+64, l+96}.
// ---------------------------------------------------------------------------
__global__ __launch_bounds__(256) void attn_kernel(
    const float* __restrict__ thx, const float* __restrict__ phx,
    const float* __restrict__ gx, __bf16* __restrict__ ybf) {
  __shared__ float sth[8][D_DIM];
  __shared__ float sg[8][D_DIM];

  const int w = threadIdx.x >> 5;
  const int l = threadIdx.x & 31;
  const int b = blockIdx.x * 8 + w;

  const float* th = thx + (size_t)b * D_DIM;
  const float* ph = phx + (size_t)b * D_DIM;
  const float* gg = gx  + (size_t)b * D_DIM;

  float t4[4], p4[4];
  float tmin = 3.4e38f, tmax = -3.4e38f;
#pragma unroll
  for (int q = 0; q < 4; ++q) {
    const int i = l + q * 32;
    t4[q] = th[i];
    p4[q] = ph[i];
    sth[w][i] = t4[q];
    sg[w][i]  = gg[i];
    tmin = fminf(tmin, t4[q]);
    tmax = fmaxf(tmax, t4[q]);
  }
  // wave32 min/max reduction of theta (for stable softmax max trick)
#pragma unroll
  for (int off = 16; off >= 1; off >>= 1) {
    tmin = fminf(tmin, __shfl_xor(tmin, off, 32));
    tmax = fmaxf(tmax, __shfl_xor(tmax, off, 32));
  }
  __syncthreads();

  float m[4], num[4], den[4];
#pragma unroll
  for (int q = 0; q < 4; ++q) {
    m[q]   = (p4[q] >= 0.0f) ? p4[q] * tmax : p4[q] * tmin;
    num[q] = 0.0f;
    den[q] = 0.0f;
  }
  for (int j = 0; j < D_DIM; ++j) {
    const float tj = sth[w][j];
    const float gj = sg[w][j];
#pragma unroll
    for (int q = 0; q < 4; ++q) {
      const float e = __expf(fmaf(p4[q], tj, -m[q]));
      num[q] = fmaf(e, gj, num[q]);
      den[q] += e;
    }
  }
#pragma unroll
  for (int q = 0; q < 4; ++q)
    ybf[(size_t)b * D_DIM + l + q * 32] = f2bf(num[q] / den[q]);
}

// ---------------------------------------------------------------------------
// Kernel 3: out = y @ W_w^T + W_b + x0.   y[B,128] bf16, W_w[1024,128].
// Block = 256 thr, 64 rows; y tile resident in LDS; 8 col-chunks x 4 K-steps.
// ---------------------------------------------------------------------------
__global__ __launch_bounds__(256) void out_kernel(
    const __bf16* __restrict__ ybf, const float* __restrict__ W_w,
    const float* __restrict__ W_b, const float* __restrict__ x0,
    float* __restrict__ out) {
  __shared__ __bf16 sy[64 * LDY];
  __shared__ __bf16 sw[128 * LDK];

  const int t  = threadIdx.x;
  const int l  = t & 31;
  const int w  = t >> 5;
  const int rt = w & 3;
  const int nh = w >> 2;
  const int m0 = blockIdx.x * 64;

  {  // stage full y tile (64 x 128) once
    const int r  = t >> 2;
    const int c0 = (t & 3) * 32;
    const __bf16* py = ybf + (size_t)(m0 + r) * D_DIM + c0;
#pragma unroll
    for (int i = 0; i < 32; ++i) sy[r * LDY + c0 + i] = py[i];
  }

  const int wr = t >> 1;
  const int wk = (t & 1) * 16;

  for (int nc = 0; nc < 8; ++nc) {  // 128 output cols per chunk
    v8f acc[4];
#pragma unroll
    for (int nt = 0; nt < 4; ++nt) acc[nt] = v8f_zero();

#pragma unroll
    for (int kc = 0; kc < D_DIM; kc += 32) {
      __syncthreads();  // also covers initial sy staging / sw reuse
      {
        const float* pw = W_w + (size_t)(nc * 128 + wr) * D_DIM + kc + wk;
#pragma unroll
        for (int i = 0; i < 16; ++i) sw[wr * LDK + wk + i] = f2bf(pw[i]);
      }
      __syncthreads();
      v16bf a = load_a_frag(sy + rt * 16 * LDY + kc, LDY);
#pragma unroll
      for (int nt = 0; nt < 4; ++nt)
        acc[nt] = wmma_bf16(a, load_b_frag(sw + (nh * 64 + nt * 16) * LDK, LDK),
                            acc[nt]);
    }

    const int colb = nc * 128 + nh * 64 + (l & 15);
    const int rowb = m0 + rt * 16 + ((l >> 4) * 8);
#pragma unroll
    for (int nt = 0; nt < 4; ++nt) {
      const int col = colb + nt * 16;
      const float bb = W_b[col];
#pragma unroll
      for (int r = 0; r < 8; ++r) {
        const size_t o = (size_t)(rowb + r) * C_DIM + col;
        out[o] = acc[nt][r] + bb + x0[o];
      }
    }
  }
}

// ---------------------------------------------------------------------------
// Host launcher
// ---------------------------------------------------------------------------
extern "C" void kernel_launch(void* const* d_in, const int* in_sizes, int n_in,
                              void* d_out, int out_size, void* d_ws, size_t ws_size,
                              hipStream_t stream) {
  (void)in_sizes; (void)n_in; (void)out_size; (void)ws_size;
  const float* x0   = (const float*)d_in[0];
  const float* x1   = (const float*)d_in[1];
  const float* g_w  = (const float*)d_in[2];
  const float* g_b  = (const float*)d_in[3];
  const float* th_w = (const float*)d_in[4];
  const float* th_b = (const float*)d_in[5];
  const float* ph_w = (const float*)d_in[6];
  const float* ph_b = (const float*)d_in[7];
  const float* W_w  = (const float*)d_in[8];
  const float* W_b  = (const float*)d_in[9];
  float* out = (float*)d_out;

  const size_t proj_bytes = (size_t)B_ROWS * D_DIM * sizeof(float);  // 8 MB
  char* ws = (char*)d_ws;
  float*  thx = (float*)(ws);
  float*  phx = (float*)(ws + proj_bytes);
  float*  gx  = (float*)(ws + 2 * proj_bytes);
  __bf16* ybf = (__bf16*)(ws + 3 * proj_bytes);

  proj_kernel<<<B_ROWS / 64, 256, 0, stream>>>(x0, x1, g_w, g_b, th_w, th_b,
                                               ph_w, ph_b, gx, thx, phx);
  attn_kernel<<<B_ROWS / 8, 256, 0, stream>>>(thx, phx, gx, ybf);
  out_kernel<<<B_ROWS / 64, 256, 0, stream>>>(ybf, W_w, W_b, x0, out);
}
